// ChainedGNN_37280316129504
// MI455X (gfx1250) — matrix operand order repaired and checked by reading, not compiled, not run on hardware
//
#include <hip/hip_runtime.h>
#include <hip/hip_bf16.h>

typedef __attribute__((ext_vector_type(16))) _Float16 v16h;
typedef __attribute__((ext_vector_type(8)))  _Float16 v8h;
typedef __attribute__((ext_vector_type(8)))  float    v8f;

#define N_NODES 512
#define N_EDGES 32768
#define HDIM    64

// ---------------------------------------------------------------- helpers
__device__ __forceinline__ v8f v8f_zero() {
  v8f v;
#pragma unroll
  for (int i = 0; i < 8; ++i) v[i] = 0.f;
  return v;
}

__device__ __forceinline__ v8f wmma16(v16h a, v16h b, v8f c) {
  return __builtin_amdgcn_wmma_f32_16x16x32_f16(false, a, false, b, (short)0, c,
                                                false, false);
}

// A fragment (16x32 f16, rows=M) from row-major [16][ld] tile at col k0.
// Lane (m=lane&15, hi=lane>>4) holds two contiguous 8-half runs:
//   k = k0 + 8*hi + i        (i = 0..7)
//   k = k0 + 16 + 8*hi + i   (i = 0..7)
// -> two 16-byte vector loads, no scalar packing.
__device__ __forceinline__ v16h load_a_frag(const _Float16* tile, int ld, int k0, int lane) {
  int m = lane & 15, hi = lane >> 4;
  const _Float16* p = tile + (size_t)m * ld + k0 + 8 * hi;
  v8h lo = *(const v8h*)(p);
  v8h hv = *(const v8h*)(p + 16);
  return __builtin_shufflevector(lo, hv, 0, 1, 2, 3, 4, 5, 6, 7,
                                 8, 9, 10, 11, 12, 13, 14, 15);
}

// B fragment from pre-packed fragment-order weights: one aligned 32B load.
// frag = kt*4 + nt ; packed[(frag*32 + lane)*16 + i].
__device__ __forceinline__ v16h load_b_packed(const _Float16* wp, int frag, int lane) {
  return *(const v16h*)(wp + (((size_t)frag * 32 + lane) << 4));
}

// ---------------------------------------------------------------- utility kernels
__global__ void zero_f32(float* p, size_t n) {
  size_t i = (size_t)blockIdx.x * blockDim.x + threadIdx.x;
  size_t stride = (size_t)gridDim.x * blockDim.x;
  for (; i < n; i += stride) p[i] = 0.f;
}

// pack f32 row-major [K][64] weight into B-fragment order (f16):
// out[((kt*4+nt)*32 + lane)*16 + i] = w[(kt*32 + 8*(lane>>4) + (i&7) + 16*(i>>3))*64 + nt*16 + (lane&15)]
__global__ void pack_b_f16(const float* __restrict__ w, _Float16* __restrict__ out, int K) {
  int p = blockIdx.x * blockDim.x + threadIdx.x;
  if (p >= K * HDIM) return;
  int i = p & 15;
  int lane = (p >> 4) & 31;
  int frag = p >> 9;
  int kt = frag >> 2, nt = frag & 3;
  int k = kt * 32 + 8 * (lane >> 4) + (i & 7) + 16 * (i >> 3);
  int n = nt * 16 + (lane & 15);
  out[p] = (_Float16)w[(size_t)k * HDIM + n];
}

// x_out[i,b] = sum_j P0[b,i,j]  -> x0[i*131 + b]
__global__ void p0_rowsum(const float* __restrict__ P0, float* __restrict__ x0) {
  int wave = threadIdx.x >> 5, lane = threadIdx.x & 31;
  int row = blockIdx.x * 8 + wave;
  int b = row >> 9, i = row & 511;
  const float* p = P0 + ((size_t)b * N_NODES + i) * N_NODES;
  float s = 0.f;
  for (int t = lane; t < N_NODES; t += 32) s += p[t];
#pragma unroll
  for (int m = 16; m >= 1; m >>= 1) s += __shfl_xor(s, m, 32);
  if (lane == 0) x0[(size_t)i * 131 + b] = s;
}

// x_in[j,b] = sum_i P0[b,i,j]  -> x0[j*131 + 64 + b]
__global__ void p0_colsum(const float* __restrict__ P0, float* __restrict__ x0) {
  int b = blockIdx.x;
  for (int j = threadIdx.x; j < N_NODES; j += blockDim.x) {
    float s = 0.f;
    for (int i = 0; i < N_NODES; ++i) s += P0[((size_t)b * N_NODES + i) * N_NODES + j];
    x0[(size_t)j * 131 + 64 + b] = s;
  }
}

__global__ void set_role(float* __restrict__ x0, const int* tx, const int* rx) {
  int i = blockIdx.x * blockDim.x + threadIdx.x;
  if (i >= N_NODES) return;
  float r0 = (i == tx[0]) ? 1.f : 0.f;
  float r1 = (i == rx[0]) ? 1.f : 0.f;
  float r2 = (r0 == 0.f && r1 == 0.f) ? 1.f : 0.f;
  x0[(size_t)i * 131 + 128] = r0;
  x0[(size_t)i * 131 + 129] = r1;
  x0[(size_t)i * 131 + 130] = r2;
}

__global__ void ln_rows_f32(const float* __restrict__ in, const float* __restrict__ g,
                            const float* __restrict__ b, float* __restrict__ out,
                            int N, int D) {
  int wave = threadIdx.x >> 5, lane = threadIdx.x & 31;
  int row = blockIdx.x * 8 + wave;
  if (row >= N) return;
  const float* x = in + (size_t)row * D;
  float s = 0.f, s2 = 0.f;
  for (int d = lane; d < D; d += 32) { float v = x[d]; s += v; s2 += v * v; }
#pragma unroll
  for (int m = 16; m >= 1; m >>= 1) { s += __shfl_xor(s, m, 32); s2 += __shfl_xor(s2, m, 32); }
  float mu = s / D, var = s2 / D - mu * mu;
  float rs = rsqrtf(var + 1e-5f);
  float* o = out + (size_t)row * D;
  for (int d = lane; d < D; d += 32) o[d] = (x[d] - mu) * rs * g[d] + b[d];
}

__global__ void ln_rows_f16(const float* __restrict__ in, const float* __restrict__ g,
                            const float* __restrict__ b, _Float16* __restrict__ out,
                            int N, int D) {
  int wave = threadIdx.x >> 5, lane = threadIdx.x & 31;
  int row = blockIdx.x * 8 + wave;
  if (row >= N) return;
  const float* x = in + (size_t)row * D;
  float s = 0.f, s2 = 0.f;
  for (int d = lane; d < D; d += 32) { float v = x[d]; s += v; s2 += v * v; }
#pragma unroll
  for (int m = 16; m >= 1; m >>= 1) { s += __shfl_xor(s, m, 32); s2 += __shfl_xor(s2, m, 32); }
  float mu = s / D, var = s2 / D - mu * mu;
  float rs = rsqrtf(var + 1e-5f);
  _Float16* o = out + (size_t)row * D;
  for (int d = lane; d < D; d += 32) o[d] = (_Float16)((x[d] - mu) * rs * g[d] + b[d]);
}

// node-side projections (tiny GEMMs commuted through the edge gather)
__global__ void node_pre_gemm(const float* __restrict__ xln, const float* __restrict__ er1w,
                              const float* __restrict__ wmsg, int ni, int ei,
                              float* __restrict__ aj, float* __restrict__ ai,
                              float* __restrict__ base) {
  int idx = blockIdx.x * blockDim.x + threadIdx.x;
  if (idx >= N_NODES * HDIM) return;
  int i = idx >> 6, o = idx & 63;
  const float* xr = xln + (size_t)i * ni;
  const float* w1 = er1w + (size_t)ei * HDIM + o;
  const float* w2 = er1w + (size_t)(ei + ni) * HDIM + o;
  const float* wm = wmsg + o;
  float s1 = 0.f, s2 = 0.f, s3 = 0.f;
  for (int k = 0; k < ni; ++k) {
    float xv = xr[k];
    s1 += xv * w1[(size_t)k * HDIM];
    s2 += xv * w2[(size_t)k * HDIM];
    s3 += xv * wm[(size_t)k * HDIM];
  }
  aj[idx] = s1; ai[idx] = s2; base[idx] = s3;
}

// ---------------------------------------------------------------- main edge kernel (WMMA)
template <int EI>
__global__ __launch_bounds__(256)
void edge_main(const _Float16* __restrict__ eln,
               const float* __restrict__ e_in,
               const int* __restrict__ src, const int* __restrict__ dst,
               const float* __restrict__ aj, const float* __restrict__ ai,
               const float* __restrict__ basev,
               const _Float16* __restrict__ w1, const float* __restrict__ b1,
               const _Float16* __restrict__ w2, const float* __restrict__ b2,
               const float* __restrict__ pg, const float* __restrict__ pb,
               const _Float16* __restrict__ wfs, const float* __restrict__ bfs,
               const _Float16* __restrict__ wfh, const float* __restrict__ bfh,
               int edge_res,
               float* __restrict__ e_out_relu, float* __restrict__ xagg) {
  __shared__ _Float16 stage[8][16][HDIM];
  int wave = threadIdx.x >> 5, lane = threadIdx.x & 31;
  int e0 = (blockIdx.x * 8 + wave) * 16;
  int nl = lane & 15, hi = lane >> 4;

  int sA[8], dA[8];
#pragma unroll
  for (int r = 0; r < 8; ++r) {
    int e = e0 + r + 8 * hi;
    sA[r] = src[e];
    dA[r] = dst[e];
  }

  // ---- GEMM1: h = leaky(e_ln @ W1 + b1 + a_j[src] + a_i[dst])
  v8f acc[4];
#pragma unroll
  for (int n = 0; n < 4; ++n) acc[n] = v8f_zero();
#pragma unroll
  for (int k0 = 0; k0 < EI; k0 += 32) {
    v16h a = load_a_frag(eln + (size_t)e0 * EI, EI, k0, lane);
#pragma unroll
    for (int n = 0; n < 4; ++n)
      acc[n] = wmma16(a, load_b_packed(w1, (k0 >> 5) * 4 + n, lane), acc[n]);
  }
#pragma unroll
  for (int n = 0; n < 4; ++n) {
    int col = n * 16 + nl;
    float bc = b1[col];
#pragma unroll
    for (int r = 0; r < 8; ++r) {
      float v = acc[n][r] + bc + aj[(size_t)sA[r] * HDIM + col] + ai[(size_t)dA[r] * HDIM + col];
      v = v > 0.f ? v : 0.1f * v;
      stage[wave][r + 8 * hi][col] = (_Float16)v;
    }
  }
  __syncthreads();

  // ---- GEMM2: e_delta = h @ W2 + b2 ; gate = sigmoid ; e_pre = gate*ed (+e)
  v8f d2[4];
#pragma unroll
  for (int n = 0; n < 4; ++n) d2[n] = v8f_zero();
#pragma unroll
  for (int k0 = 0; k0 < 64; k0 += 32) {
    v16h a = load_a_frag(&stage[wave][0][0], HDIM, k0, lane);
#pragma unroll
    for (int n = 0; n < 4; ++n)
      d2[n] = wmma16(a, load_b_packed(w2, (k0 >> 5) * 4 + n, lane), d2[n]);
  }
  float ep[4][8];
#pragma unroll
  for (int n = 0; n < 4; ++n) {
    int col = n * 16 + nl;
    float bc = b2[col];
#pragma unroll
    for (int r = 0; r < 8; ++r) {
      float ed = d2[n][r] + bc;
      float gate = 1.f / (1.f + expf(-ed));
      float v = gate * ed;
      if (edge_res) v += e_in[(size_t)(e0 + r + 8 * hi) * HDIM + col];
      ep[n][r] = v;
    }
  }

  // ---- epost layernorm per edge row (16 lanes of one half-wave hold one row)
  float mu[8], rs[8];
#pragma unroll
  for (int r = 0; r < 8; ++r) {
    float s = ep[0][r] + ep[1][r] + ep[2][r] + ep[3][r];
    float s2 = ep[0][r] * ep[0][r] + ep[1][r] * ep[1][r] + ep[2][r] * ep[2][r] +
               ep[3][r] * ep[3][r];
#pragma unroll
    for (int m = 8; m >= 1; m >>= 1) { s += __shfl_xor(s, m, 16); s2 += __shfl_xor(s2, m, 16); }
    float m_ = s * (1.f / 64.f);
    mu[r] = m_;
    rs[r] = rsqrtf(s2 * (1.f / 64.f) - m_ * m_ + 1e-5f);
  }
  __syncthreads();
#pragma unroll
  for (int n = 0; n < 4; ++n) {
    int col = n * 16 + nl;
    float gc = pg[col], bc = pb[col];
#pragma unroll
    for (int r = 0; r < 8; ++r) {
      float eo = (ep[n][r] - mu[r]) * rs[r] * gc + bc;
      e_out_relu[(size_t)(e0 + r + 8 * hi) * HDIM + col] = fmaxf(eo, 0.f);
      stage[wave][r + 8 * hi][col] = (_Float16)eo;   // pre-relu for fs/fh
    }
  }
  __syncthreads();

  // ---- GEMM3/4: scale/shift ; msg = (1+scale)*base[src] + shift ; atomic scatter
  v8f cs[4], ch[4];
#pragma unroll
  for (int n = 0; n < 4; ++n) { cs[n] = v8f_zero(); ch[n] = v8f_zero(); }
#pragma unroll
  for (int k0 = 0; k0 < 64; k0 += 32) {
    v16h a = load_a_frag(&stage[wave][0][0], HDIM, k0, lane);
#pragma unroll
    for (int n = 0; n < 4; ++n) {
      cs[n] = wmma16(a, load_b_packed(wfs, (k0 >> 5) * 4 + n, lane), cs[n]);
      ch[n] = wmma16(a, load_b_packed(wfh, (k0 >> 5) * 4 + n, lane), ch[n]);
    }
  }
#pragma unroll
  for (int n = 0; n < 4; ++n) {
    int col = n * 16 + nl;
    float fsb = bfs[col], fhb = bfh[col];
#pragma unroll
    for (int r = 0; r < 8; ++r) {
      float msg = (1.f + cs[n][r] + fsb) * basev[(size_t)sA[r] * HDIM + col] +
                  (ch[n][r] + fhb);
      atomicAdd(&xagg[(size_t)dA[r] * HDIM + col], msg);
    }
  }
}

// node FF + residual + LN + relu. One block (64 threads) per node.
__global__ void node_update(const float* __restrict__ xagg, const float* __restrict__ xprev,
                            const float* __restrict__ ff1w, const float* __restrict__ ff1b,
                            const float* __restrict__ ff2w, const float* __restrict__ ff2b,
                            const float* __restrict__ g, const float* __restrict__ b,
                            int node_res, float* __restrict__ xs_out) {
  __shared__ float h2[HDIM];
  __shared__ float up[HDIM];
  int i = blockIdx.x, o = threadIdx.x;
  float s = ff1b[o];
  for (int k = 0; k < HDIM; ++k) s += xagg[(size_t)i * HDIM + k] * ff1w[(size_t)k * HDIM + o];
  s = s > 0.f ? s : 0.1f * s;
  h2[o] = s;
  __syncthreads();
  float u = ff2b[o];
  for (int k = 0; k < HDIM; ++k) u += h2[k] * ff2w[(size_t)k * HDIM + o];
  if (node_res) u += xprev[(size_t)i * HDIM + o];
  up[o] = u;
  __syncthreads();
  float mu = 0.f, s2 = 0.f;
  for (int k = 0; k < HDIM; ++k) { mu += up[k]; s2 += up[k] * up[k]; }
  mu *= (1.f / 64.f);
  float var = s2 * (1.f / 64.f) - mu * mu;
  float v = (u - mu) * rsqrtf(var + 1e-5f) * g[o] + b[o];
  xs_out[(size_t)i * HDIM + o] = fmaxf(v, 0.f);
}

// JK projection
__global__ void jk_proj(const float* __restrict__ xs0, const float* __restrict__ xs1,
                        const float* __restrict__ xs2, int nl,
                        const float* __restrict__ w, const float* __restrict__ bb,
                        float* __restrict__ xp) {
  int idx = blockIdx.x * blockDim.x + threadIdx.x;
  if (idx >= N_NODES * HDIM) return;
  int i = idx >> 6, o = idx & 63;
  const float* xsv[3] = {xs0, xs1, xs2};
  float s = bb[o];
  for (int ll = 0; ll < nl; ++ll)
    for (int k = 0; k < HDIM; ++k)
      s += xsv[ll][(size_t)i * HDIM + k] * w[(size_t)(ll * HDIM + k) * HDIM + o];
  xp[idx] = s;
}

// dec_in = concat(es, xp[src], xp[dst]), layernorm over 192, f16 out.
__global__ void dec_ln(const float* __restrict__ es, const float* __restrict__ xp,
                       const int* __restrict__ src, const int* __restrict__ dst,
                       const float* __restrict__ g, const float* __restrict__ b,
                       _Float16* __restrict__ out) {
  int wave = threadIdx.x >> 5, lane = threadIdx.x & 31;
  int e = blockIdx.x * 8 + wave;
  if (e >= N_EDGES) return;
  int sj = src[e], di = dst[e];
  float vals[6];
  float s = 0.f, s2 = 0.f;
#pragma unroll
  for (int t = 0; t < 6; ++t) {
    int d = lane + 32 * t;
    float v = (d < 64) ? es[(size_t)e * 64 + d]
              : (d < 128) ? xp[(size_t)sj * 64 + (d - 64)]
                          : xp[(size_t)di * 64 + (d - 128)];
    vals[t] = v; s += v; s2 += v * v;
  }
#pragma unroll
  for (int m = 16; m >= 1; m >>= 1) { s += __shfl_xor(s, m, 32); s2 += __shfl_xor(s2, m, 32); }
  float mu = s * (1.f / 192.f);
  float rs = rsqrtf(s2 * (1.f / 192.f) - mu * mu + 1e-5f);
#pragma unroll
  for (int t = 0; t < 6; ++t) {
    int d = lane + 32 * t;
    out[(size_t)e * 192 + d] = (_Float16)((vals[t] - mu) * rs * g[d] + b[d]);
  }
}

// decoder: p = softplus(dec_ln @ head_w + head_b); scatter into out[b, src, dst].
__global__ __launch_bounds__(256)
void decoder(const _Float16* __restrict__ dec16, const _Float16* __restrict__ hw,
             const float* __restrict__ hb, const int* __restrict__ src,
             const int* __restrict__ dst, float* __restrict__ out) {
  int wave = threadIdx.x >> 5, lane = threadIdx.x & 31;
  int e0 = (blockIdx.x * 8 + wave) * 16;
  int nl = lane & 15, hi = lane >> 4;
  v8f acc[4];
#pragma unroll
  for (int n = 0; n < 4; ++n) acc[n] = v8f_zero();
#pragma unroll
  for (int k0 = 0; k0 < 192; k0 += 32) {
    v16h a = load_a_frag(dec16 + (size_t)e0 * 192, 192, k0, lane);
#pragma unroll
    for (int n = 0; n < 4; ++n)
      acc[n] = wmma16(a, load_b_packed(hw, (k0 >> 5) * 4 + n, lane), acc[n]);
  }
#pragma unroll
  for (int r = 0; r < 8; ++r) {
    int e = e0 + r + 8 * hi;
    size_t off = (size_t)src[e] * N_NODES + dst[e];
#pragma unroll
    for (int n = 0; n < 4; ++n) {
      int col = n * 16 + nl;
      float v = acc[n][r] + hb[col];
      float sp = fmaxf(v, 0.f) + log1pf(expf(-fabsf(v)));   // stable softplus
      out[(size_t)col * ((size_t)N_NODES * N_NODES) + off] = sp;
    }
  }
}

// ---------------------------------------------------------------- launch
extern "C" void kernel_launch(void* const* d_in, const int* in_sizes, int n_in,
                              void* d_out, int out_size, void* d_ws, size_t ws_size,
                              hipStream_t stream) {
  (void)in_sizes; (void)n_in; (void)ws_size;
  // pytree (sorted-key) input order:
  // 0 P0, 1 edge_index, 2 edge_attr, 3 dec_b, 4 dec_g, 5 head_b, 6 head_w,
  // 7..12 jk{0..2}{b,w}, 13..75 layers (21 leaves each, alpha order), 76 tx, 77 rx
  const float* P0    = (const float*)d_in[0];
  const int*   eidx  = (const int*)d_in[1];
  const int*   src   = eidx;
  const int*   dst   = eidx + N_EDGES;
  const float* eattr = (const float*)d_in[2];
  const float* dec_b = (const float*)d_in[3];
  const float* dec_g = (const float*)d_in[4];
  const float* head_b = (const float*)d_in[5];
  const float* head_w = (const float*)d_in[6];
  const float* jk_b[3] = {(const float*)d_in[7], (const float*)d_in[9], (const float*)d_in[11]};
  const float* jk_w[3] = {(const float*)d_in[8], (const float*)d_in[10], (const float*)d_in[12]};
  const int* tx = (const int*)d_in[76];
  const int* rx = (const int*)d_in[77];

  char* ws = (char*)d_ws;
  size_t off = 0;
  auto carve = [&](size_t bytes) -> void* {
    off = (off + 255) & ~(size_t)255;
    void* p = ws + off;
    off += bytes;
    return p;
  };
  float* x0    = (float*)carve((size_t)N_NODES * 131 * 4);
  float* xln   = (float*)carve((size_t)N_NODES * 131 * 4);
  float* aj    = (float*)carve((size_t)N_NODES * HDIM * 4);
  float* ai    = (float*)carve((size_t)N_NODES * HDIM * 4);
  float* basev = (float*)carve((size_t)N_NODES * HDIM * 4);
  float* xagg  = (float*)carve((size_t)N_NODES * HDIM * 4);
  float* xs[3];
  for (int l = 0; l < 3; ++l) xs[l] = (float*)carve((size_t)N_NODES * HDIM * 4);
  float* xproj = (float*)carve((size_t)N_NODES * HDIM * 4);
  float* ebuf[2];
  for (int i = 0; i < 2; ++i) ebuf[i] = (float*)carve((size_t)N_EDGES * HDIM * 4);
  _Float16* f16buf  = (_Float16*)carve((size_t)N_EDGES * 192 * 2);  // e_ln / dec_ln
  _Float16* w16_er1 = (_Float16*)carve((size_t)128 * HDIM * 2);
  _Float16* w16_er2 = (_Float16*)carve((size_t)64 * HDIM * 2);
  _Float16* w16_fs  = (_Float16*)carve((size_t)64 * HDIM * 2);
  _Float16* w16_fh  = (_Float16*)carve((size_t)64 * HDIM * 2);
  _Float16* w16_hd  = (_Float16*)carve((size_t)192 * HDIM * 2);

  zero_f32<<<8192, 256, 0, stream>>>((float*)d_out, (size_t)out_size);
  p0_rowsum<<<4096, 256, 0, stream>>>(P0, x0);
  p0_colsum<<<64, 256, 0, stream>>>(P0, x0);
  set_role<<<2, 256, 0, stream>>>(x0, tx, rx);
  pack_b_f16<<<48, 256, 0, stream>>>(head_w, w16_hd, 192);

  const float* ecur = eattr;
  for (int l = 0; l < 3; ++l) {
    int bi = 13 + l * 21;
    const float* epost_b = (const float*)d_in[bi + 0];
    const float* epost_g = (const float*)d_in[bi + 1];
    const float* epre_b  = (const float*)d_in[bi + 2];
    const float* epre_g  = (const float*)d_in[bi + 3];
    const float* er1_b   = (const float*)d_in[bi + 4];
    const float* er1_w   = (const float*)d_in[bi + 5];
    const float* er2_b   = (const float*)d_in[bi + 6];
    const float* er2_w   = (const float*)d_in[bi + 7];
    const float* ff1_b   = (const float*)d_in[bi + 8];
    const float* ff1_w   = (const float*)d_in[bi + 9];
    const float* ff2_b   = (const float*)d_in[bi + 10];
    const float* ff2_w   = (const float*)d_in[bi + 11];
    const float* fh_b    = (const float*)d_in[bi + 12];
    const float* fh_w    = (const float*)d_in[bi + 13];
    const float* fs_b    = (const float*)d_in[bi + 14];
    const float* fs_w    = (const float*)d_in[bi + 15];
    const float* npost_b = (const float*)d_in[bi + 16];
    const float* npost_g = (const float*)d_in[bi + 17];
    const float* npre_b  = (const float*)d_in[bi + 18];
    const float* npre_g  = (const float*)d_in[bi + 19];
    const float* w_msg   = (const float*)d_in[bi + 20];

    int ni = (l == 0) ? 131 : 64;
    int ei = (l == 0) ? 128 : 64;
    const float* xin = (l == 0) ? x0 : xs[l - 1];
    float* eout = ebuf[l & 1];

    zero_f32<<<32, 256, 0, stream>>>(xagg, (size_t)N_NODES * HDIM);
    ln_rows_f32<<<64, 256, 0, stream>>>(xin, npre_g, npre_b, xln, N_NODES, ni);
    node_pre_gemm<<<128, 256, 0, stream>>>(xln, er1_w, w_msg, ni, ei, aj, ai, basev);
    pack_b_f16<<<(ei * HDIM + 255) / 256, 256, 0, stream>>>(er1_w, w16_er1, ei);
    pack_b_f16<<<16, 256, 0, stream>>>(er2_w, w16_er2, 64);
    pack_b_f16<<<16, 256, 0, stream>>>(fs_w, w16_fs, 64);
    pack_b_f16<<<16, 256, 0, stream>>>(fh_w, w16_fh, 64);
    ln_rows_f16<<<4096, 256, 0, stream>>>(ecur, epre_g, epre_b, f16buf, N_EDGES, ei);
    if (l == 0) {
      edge_main<128><<<256, 256, 0, stream>>>(f16buf, ecur, src, dst, aj, ai, basev,
                                              w16_er1, er1_b, w16_er2, er2_b,
                                              epost_g, epost_b, w16_fs, fs_b,
                                              w16_fh, fh_b, 0, eout, xagg);
    } else {
      edge_main<64><<<256, 256, 0, stream>>>(f16buf, ecur, src, dst, aj, ai, basev,
                                             w16_er1, er1_b, w16_er2, er2_b,
                                             epost_g, epost_b, w16_fs, fs_b,
                                             w16_fh, fh_b, 1, eout, xagg);
    }
    node_update<<<512, 64, 0, stream>>>(xagg, (l > 0) ? xs[l - 1] : xs[0],
                                        ff1_w, ff1_b, ff2_w, ff2_b,
                                        npost_g, npost_b, (l > 0) ? 1 : 0, xs[l]);
    jk_proj<<<128, 256, 0, stream>>>(xs[0], xs[1], xs[2], l + 1, jk_w[l], jk_b[l], xproj);
    dec_ln<<<4096, 256, 0, stream>>>(eout, xproj, src, dst, dec_g, dec_b, f16buf);
    decoder<<<256, 256, 0, stream>>>(f16buf, w16_hd, head_b, src, dst,
                                     (float*)d_out + (size_t)l * 64 * N_NODES * N_NODES);
    ecur = eout;
  }
}
